// MambaSSMBlock_69458211111446
// MI455X (gfx1250) — matrix operand order repaired and checked by reading, not compiled, not run on hardware
//
#include <hip/hip_runtime.h>
#include <hip/hip_bf16.h>
#include <cstdint>
#include <cstddef>

// ---------------- problem constants (from reference) ----------------
#define HID    1024
#define STATE  16
#define KCONV  4
#define INNER  2048            // EXPAND * HID
#define BSZ    4
#define LSEQ   4096
#define MROWS  (BSZ * LSEQ)    // 16384
#define N1     (2 * INNER)     // 4096

typedef __bf16 bf16;
typedef __attribute__((ext_vector_type(16))) __bf16 v16bf;
typedef __attribute__((ext_vector_type(8)))  __bf16 v8bf;
typedef __attribute__((ext_vector_type(8)))  float  v8f;

// ---------------- small math helpers ----------------
__device__ __forceinline__ float siluf(float v)     { return v / (1.0f + __expf(-v)); }
__device__ __forceinline__ float softplusf(float v) { return v > 20.0f ? v : log1pf(__expf(v)); }

__device__ __forceinline__ v16bf make16(v8bf lo, v8bf hi) {
    union { v16bf v; struct { v8bf lo, hi; } s; } u;
    u.s.lo = lo; u.s.hi = hi;
    return u.v;
}

// ---------------- elementwise cast f32 -> bf16 ----------------
__global__ __launch_bounds__(256) void cast_f32_bf16(const float* __restrict__ in,
                                                     bf16* __restrict__ out, size_t n) {
    size_t i = (size_t)blockIdx.x * blockDim.x + threadIdx.x;
    if (i < n) out[i] = (bf16)in[i];
}

// ---------------- copy f32 (rows x cols) -> bf16 (rows x cols), layout unchanged ----------------
__global__ __launch_bounds__(256) void transpose_to_bf16(const float* __restrict__ in,
                                                         bf16* __restrict__ out,
                                                         int rows, int cols) {
    size_t i = (size_t)blockIdx.x * blockDim.x + threadIdx.x;
    size_t n = (size_t)rows * cols;
    if (i >= n) return;
    int r = (int)(i / cols);
    int c = (int)(i % cols);
    out[(size_t)r * cols + c] = (bf16)in[i];
}

// ---------------- WMMA GEMM: C[M,N] = A[M,K](bf16) * Bt[N,K]^T(bf16) + bias, epilogue ---
// Workgroup tile 128x128; 8 waves, each wave owns 32(M) x 64(N) via 2x4 16x16 accumulators.
// Fragment layouts per CDNA5 ISA 7.12.2:
//   A 16x32: lanes 0-15 -> M rows with K 0..7 & 16..23 ; lanes 16-31 -> K 8..15 & 24..31
//   B 32x16 (from Bt stored N-major): lanes 0-15 -> N cols with K 0..7 & 16..23 ; lanes 16-31 -> K 8..15 & 24..31
// EPI 0: bias -> C (ldc = N)
// EPI 1: split at N/2: cols [0,N/2) bias->C ; cols [N/2,N) bias+SiLU->C2 (both ld = N/2)
// EPI 2: bias + softplus -> C (ldc = N)
template <int EPI>
__global__ __launch_bounds__(256) void gemm_wmma_bf16(const bf16* __restrict__ A,
                                                      const bf16* __restrict__ Bt,
                                                      const float* __restrict__ bias,
                                                      float* __restrict__ C,
                                                      float* __restrict__ C2,
                                                      int M, int N, int K) {
    const int lane  = threadIdx.x & 31;
    const int wave  = threadIdx.x >> 5;
    const int tileM = blockIdx.y * 128 + (wave >> 1) * 32;
    const int tileN = blockIdx.x * 128 + (wave & 1) * 64;

    v8f acc[2][4] = {};

    const int kb = (lane >> 4) << 3;   // 0 or 8
    const bf16* pa0 = A + (size_t)(tileM + (lane & 15)) * K + kb;
    const bf16* pa1 = pa0 + (size_t)16 * K;
    const bf16* pb  = Bt + (size_t)(tileN + (lane & 15)) * K + kb;
    const size_t bstep = (size_t)32;

    for (int k = 0; k < K; k += 32) {
        // speculative prefetch of next K-slice (global_prefetch_b8; OOB is dropped)
        __builtin_prefetch(pa0 + 32, 0, 1);
        __builtin_prefetch(pa1 + 32, 0, 1);
        __builtin_prefetch(pb + bstep, 0, 1);

        v16bf a0 = make16(*(const v8bf*)pa0, *(const v8bf*)(pa0 + 16));
        v16bf a1 = make16(*(const v8bf*)pa1, *(const v8bf*)(pa1 + 16));
        v16bf b0 = make16(*(const v8bf*)(pb), *(const v8bf*)(pb + 16));
        v16bf b1 = make16(*(const v8bf*)(pb + (size_t)16 * K), *(const v8bf*)(pb + (size_t)16 * K + 16));
        v16bf b2 = make16(*(const v8bf*)(pb + (size_t)32 * K), *(const v8bf*)(pb + (size_t)32 * K + 16));
        v16bf b3 = make16(*(const v8bf*)(pb + (size_t)48 * K), *(const v8bf*)(pb + (size_t)48 * K + 16));

        acc[0][0] = __builtin_amdgcn_wmma_f32_16x16x32_bf16(false, a0, false, b0, (short)0, acc[0][0], false, false);
        acc[0][1] = __builtin_amdgcn_wmma_f32_16x16x32_bf16(false, a0, false, b1, (short)0, acc[0][1], false, false);
        acc[0][2] = __builtin_amdgcn_wmma_f32_16x16x32_bf16(false, a0, false, b2, (short)0, acc[0][2], false, false);
        acc[0][3] = __builtin_amdgcn_wmma_f32_16x16x32_bf16(false, a0, false, b3, (short)0, acc[0][3], false, false);
        acc[1][0] = __builtin_amdgcn_wmma_f32_16x16x32_bf16(false, a1, false, b0, (short)0, acc[1][0], false, false);
        acc[1][1] = __builtin_amdgcn_wmma_f32_16x16x32_bf16(false, a1, false, b1, (short)0, acc[1][1], false, false);
        acc[1][2] = __builtin_amdgcn_wmma_f32_16x16x32_bf16(false, a1, false, b2, (short)0, acc[1][2], false, false);
        acc[1][3] = __builtin_amdgcn_wmma_f32_16x16x32_bf16(false, a1, false, b3, (short)0, acc[1][3], false, false);

        pa0 += 32; pa1 += 32; pb += bstep;
    }

    // C/D layout: lanes 0-15 -> N=lane, M = m0 + r ; lanes 16-31 -> N=lane-16, M = m0 + 8 + r
    const int ncol = lane & 15;
    const int mofs = (lane >> 4) << 3;
    const int Nh   = N >> 1;

    #pragma unroll
    for (int i = 0; i < 2; ++i) {
        #pragma unroll
        for (int j = 0; j < 4; ++j) {
            const int mbase = tileM + (i << 4) + mofs;
            const int n     = tileN + (j << 4) + ncol;
            const float bv  = bias[n];
            #pragma unroll
            for (int r = 0; r < 8; ++r) {
                float v = acc[i][j][r] + bv;
                size_t row = (size_t)(mbase + r);
                if (EPI == 0) {
                    C[row * N + n] = v;
                } else if (EPI == 1) {
                    if (n < Nh) C [row * Nh + n]        = v;
                    else        C2[row * Nh + (n - Nh)] = siluf(v);
                } else { // EPI == 2
                    C[row * N + n] = softplusf(v);
                }
            }
        }
    }
}

// ---------------- causal depthwise conv (K=4) + SiLU, emit f32 and bf16 ----------------
__global__ __launch_bounds__(256) void conv_silu(const float* __restrict__ xpart,
                                                 const float* __restrict__ conv_w,
                                                 const float* __restrict__ conv_b,
                                                 float* __restrict__ xconv,
                                                 bf16*  __restrict__ xconv_bf) {
    size_t idx = (size_t)blockIdx.x * blockDim.x + threadIdx.x;
    if (idx >= (size_t)MROWS * INNER) return;
    int c = (int)(idx % INNER);
    int l = (int)((idx / INNER) % LSEQ);
    int b = (int)(idx / ((size_t)INNER * LSEQ));
    float acc = conv_b[c];
    #pragma unroll
    for (int j = 0; j < KCONV; ++j) {
        int li = l - (KCONV - 1) + j;
        if (li >= 0)
            acc += conv_w[c * KCONV + j] * xpart[((size_t)b * LSEQ + li) * INNER + c];
    }
    float s = siluf(acc);
    xconv[idx]    = s;
    xconv_bf[idx] = (bf16)s;
}

// ---------------- B/C projections: N=16 each, one wave per row ------------------------
// Row staged memory -> LDS via GLOBAL_LOAD_ASYNC_TO_LDS_B128 (ASYNCcnt path, no VGPR bounce)
__global__ __launch_bounds__(128) void bc_proj(const float* __restrict__ xconv,
                                               const float* __restrict__ Bw,
                                               const float* __restrict__ Bb,
                                               const float* __restrict__ Cw,
                                               const float* __restrict__ Cb,
                                               float* __restrict__ Bm,
                                               float* __restrict__ Cm) {
    __shared__ float srow[4][INNER];       // 4 waves x 2048 f32 = 32 KB of the 320 KB WGP LDS
    const int lane = threadIdx.x & 31;
    const int wave = threadIdx.x >> 5;
    const int row  = blockIdx.x * 4 + wave;

    // async-copy this wave's row: 2048 f32 = 8192 B, 16 B per lane per issue, 16 issues
    {
        uint64_t gbase = (uint64_t)(const void*)(xconv + (size_t)row * INNER);
        uint32_t lbase = (uint32_t)(uintptr_t)(&srow[wave][0]);
        #pragma unroll
        for (int it = 0; it < INNER / (32 * 4); ++it) {
            uint32_t byteoff = (uint32_t)(it * 32 + lane) * 16u;
            uint64_t ga = gbase + byteoff;
            uint32_t la = lbase + byteoff;
            asm volatile("global_load_async_to_lds_b128 %0, %1, off"
                         :: "v"(la), "v"(ga) : "memory");
        }
        asm volatile("s_wait_asynccnt 0x0" ::: "memory");
    }
    __syncthreads();

    const float* W = (lane < 16) ? Bw : Cw;
    const int s = lane & 15;
    float acc = 0.0f;
    for (int k = 0; k < INNER; ++k)
        acc += srow[wave][k] * W[(size_t)s * INNER + k];
    acc += (lane < 16 ? Bb : Cb)[s];
    float* Out = (lane < 16) ? Bm : Cm;
    Out[(size_t)row * STATE + s] = acc;
}

// ---------------- selective scan, fused SiLU(z) gating + bf16 downcast -----------------
// grid: (INNER/256, BSZ); thread owns one channel, h[16] in registers, L sequential.
// Per-timestep B/C broadcast staged via GLOBAL_LOAD_ASYNC_TO_LDS_B32.
__global__ __launch_bounds__(256) void selective_scan(const float* __restrict__ xconv,
                                                      const float* __restrict__ dt,
                                                      const float* __restrict__ A_log,
                                                      const float* __restrict__ Bm,
                                                      const float* __restrict__ Cm,
                                                      const float* __restrict__ Dv,
                                                      const float* __restrict__ zsilu,
                                                      bf16* __restrict__ ybf) {
    const int c = blockIdx.x * 256 + threadIdx.x;
    const int b = blockIdx.y;
    float A[STATE];
    #pragma unroll
    for (int s = 0; s < STATE; ++s) A[s] = -__expf(A_log[(size_t)c * STATE + s]);
    const float Dc = Dv[c];
    float h[STATE];
    #pragma unroll
    for (int s = 0; s < STATE; ++s) h[s] = 0.0f;

    __shared__ float sBC[2 * STATE];
    const uint32_t lbc = (uint32_t)(uintptr_t)(&sBC[0]);

    for (int t = 0; t < LSEQ; ++t) {
        const size_t row = (size_t)b * LSEQ + t;
        if (threadIdx.x < 2 * STATE) {
            const float* src = (threadIdx.x < STATE)
                ? (Bm + row * STATE + threadIdx.x)
                : (Cm + row * STATE + (threadIdx.x - STATE));
            uint64_t ga = (uint64_t)(const void*)src;
            uint32_t la = lbc + threadIdx.x * 4u;
            asm volatile("global_load_async_to_lds_b32 %0, %1, off"
                         :: "v"(la), "v"(ga) : "memory");
            asm volatile("s_wait_asynccnt 0x0" ::: "memory");
        }
        __syncthreads();
        const size_t idx = row * INNER + c;
        const float xv  = xconv[idx];
        const float dtv = dt[idx];
        float y = Dc * xv;
        #pragma unroll
        for (int s = 0; s < STATE; ++s) {
            float dA = __expf(dtv * A[s]);
            h[s] = dA * h[s] + dtv * sBC[s] * xv;
            y += h[s] * sBC[STATE + s];
        }
        ybf[idx] = (bf16)(y * zsilu[idx]);
        __syncthreads();
    }
}

// ---------------- host-side launcher ----------------
static inline size_t align256(size_t v) { return (v + 255) & ~(size_t)255; }

extern "C" void kernel_launch(void* const* d_in, const int* in_sizes, int n_in,
                              void* d_out, int out_size, void* d_ws, size_t ws_size,
                              hipStream_t stream) {
    (void)in_sizes; (void)n_in; (void)out_size; (void)ws_size;

    const float* x         = (const float*)d_in[0];
    const float* in_proj_w = (const float*)d_in[1];
    const float* in_proj_b = (const float*)d_in[2];
    const float* conv_w    = (const float*)d_in[3];
    const float* conv_b    = (const float*)d_in[4];
    const float* dt_w      = (const float*)d_in[5];
    const float* dt_b      = (const float*)d_in[6];
    const float* A_log     = (const float*)d_in[7];
    const float* B_w       = (const float*)d_in[8];
    const float* B_b       = (const float*)d_in[9];
    const float* C_w       = (const float*)d_in[10];
    const float* C_b       = (const float*)d_in[11];
    const float* Dv        = (const float*)d_in[12];
    const float* out_w     = (const float*)d_in[13];
    const float* out_b     = (const float*)d_in[14];
    float* out             = (float*)d_out;

    // workspace carve-up
    char* p = (char*)d_ws;
    size_t off = 0;
    auto carve = [&](size_t bytes) { char* q = p + off; off = align256(off + bytes); return q; };
    bf16*  x_bf     = (bf16*) carve((size_t)MROWS * HID   * sizeof(bf16));
    bf16*  wt_in    = (bf16*) carve((size_t)HID   * N1    * sizeof(bf16));   // (N=4096, K=HID)
    bf16*  wt_dt    = (bf16*) carve((size_t)INNER * INNER * sizeof(bf16));   // (N, K) = (2048,2048)
    bf16*  wt_out   = (bf16*) carve((size_t)INNER * HID   * sizeof(bf16));   // (N=1024, K=2048)
    float* xpart    = (float*)carve((size_t)MROWS * INNER * sizeof(float));
    float* zsilu    = (float*)carve((size_t)MROWS * INNER * sizeof(float));
    float* xconv    = (float*)carve((size_t)MROWS * INNER * sizeof(float));
    bf16*  xconv_bf = (bf16*) carve((size_t)MROWS * INNER * sizeof(bf16));
    float* dtbuf    = (float*)carve((size_t)MROWS * INNER * sizeof(float));
    float* Bm       = (float*)carve((size_t)MROWS * STATE * sizeof(float));
    float* Cm       = (float*)carve((size_t)MROWS * STATE * sizeof(float));
    bf16*  ybf      = (bf16*) carve((size_t)MROWS * INNER * sizeof(bf16));

    // 1) precision conversion / weight casts (one-shot, deterministic)
    {
        size_t n = (size_t)MROWS * HID;
        cast_f32_bf16<<<(unsigned)((n + 255) / 256), 256, 0, stream>>>(x, x_bf, n);
    }
    {
        size_t n = (size_t)N1 * HID;
        transpose_to_bf16<<<(unsigned)((n + 255) / 256), 256, 0, stream>>>(in_proj_w, wt_in, N1, HID);
    }
    {
        size_t n = (size_t)INNER * INNER;
        transpose_to_bf16<<<(unsigned)((n + 255) / 256), 256, 0, stream>>>(dt_w, wt_dt, INNER, INNER);
    }
    {
        size_t n = (size_t)HID * INNER;
        transpose_to_bf16<<<(unsigned)((n + 255) / 256), 256, 0, stream>>>(out_w, wt_out, HID, INNER);
    }

    // 2) in_proj GEMM: xz = x @ W_in^T + b ; split -> xpart (plain) + zsilu (SiLU applied)
    gemm_wmma_bf16<1><<<dim3(N1 / 128, MROWS / 128), 256, 0, stream>>>(
        x_bf, wt_in, in_proj_b, xpart, zsilu, MROWS, N1, HID);

    // 3) causal depthwise conv + SiLU
    {
        size_t n = (size_t)MROWS * INNER;
        conv_silu<<<(unsigned)((n + 255) / 256), 256, 0, stream>>>(xpart, conv_w, conv_b, xconv, xconv_bf);
    }

    // 4) dt GEMM with fused softplus
    gemm_wmma_bf16<2><<<dim3(INNER / 128, MROWS / 128), 256, 0, stream>>>(
        xconv_bf, wt_dt, dt_b, dtbuf, nullptr, MROWS, INNER, INNER);

    // 5) B/C projections (N=16 each)
    bc_proj<<<MROWS / 4, 128, 0, stream>>>(xconv, B_w, B_b, C_w, C_b, Bm, Cm);

    // 6) selective scan, fused gate + bf16 cast
    selective_scan<<<dim3(INNER / 256, BSZ), 256, 0, stream>>>(
        xconv, dtbuf, A_log, Bm, Cm, Dv, zsilu, ybf);

    // 7) output GEMM straight into d_out
    gemm_wmma_bf16<0><<<dim3(HID / 128, MROWS / 128), 256, 0, stream>>>(
        ybf, wt_out, out_b, out, nullptr, MROWS, HID, INNER);
}
